// Encoder_39127152067066
// MI455X (gfx1250) — matrix operand order repaired and checked by reading, not compiled, not run on hardware
//
#include <hip/hip_runtime.h>

// ---------------------------------------------------------------------------
// Bidirectional GRU encoder for MI455X (gfx1250, wave32, WMMA + TDM).
//   V=32000, E=512, H=1024, B=64, T=512
//   1) embed gather -> bf16
//   2) weights fp32 -> bf16 once
//   3) xp = emb @ W_ih^T + b_ih (both dirs): WMMA GEMM, W tile (64KB) staged
//      in LDS via TENSOR_LOAD_TO_LDS (TDM), B-frags from LDS.
//   4) persistent scan: 64 WGs/dir own a 16-wide hidden tile; W_hh slice
//      (3x16x1024 bf16 = 96KB) TDM-staged in LDS once; per step 3 gate tiles
//      x 4 M-tiles of v_wmma_f32_16x16x32_bf16; f32 gates; device-scope
//      atomic grid barrier per direction.
//   5) hiddens = tanh([h_f,h_b] @ fc_W^T + b): small WMMA GEMM.
// ---------------------------------------------------------------------------

typedef __bf16 bf16_t;
typedef __attribute__((ext_vector_type(16))) __bf16 v16bf;
typedef __attribute__((ext_vector_type(8)))  __bf16 v8bf;
typedef __attribute__((ext_vector_type(8)))  float  v8f;
typedef __attribute__((ext_vector_type(4)))  unsigned int u32x4;
typedef __attribute__((ext_vector_type(8)))  int i32x8;
typedef __attribute__((ext_vector_type(4)))  int i32x4;

#define BB   64
#define TT   512
#define BT   32768      // B*T
#define EE   512
#define HH   1024
#define HH2  2048
#define HH3  3072

#if __has_builtin(__builtin_amdgcn_tensor_load_to_lds) && \
    __has_builtin(__builtin_amdgcn_s_wait_tensorcnt)
#define USE_TDM 1
#else
#define USE_TDM 0
#endif

__device__ __forceinline__ bf16_t f2bf(float f) {
  unsigned u = __float_as_uint(f);
  u += 0x7FFFu + ((u >> 16) & 1u);          // round-to-nearest-even
  unsigned short s = (unsigned short)(u >> 16);
  bf16_t r; __builtin_memcpy(&r, &s, 2); return r;
}

__device__ __forceinline__ v16bf cat8(v8bf lo, v8bf hi) {
  return __builtin_shufflevector(lo, hi, 0,1,2,3,4,5,6,7,8,9,10,11,12,13,14,15);
}

// A fragment (16x32 bf16) from row-major src at (row0,k0); lane m -> row m&15,
// K chunks at +8*(m>>4) and +16 more.  (ISA 7.12.2, 16-bit A 16x32)
__device__ __forceinline__ v16bf frag_a(const bf16_t* base, int lda, int lane) {
  const bf16_t* p = base + (size_t)(lane & 15) * lda + ((lane >> 4) * 8);
  v8bf lo = *(const v8bf*)(p);
  v8bf hi = *(const v8bf*)(p + 16);
  return cat8(lo, hi);
}

// B fragment (32x16) where B[k][n] = W[n][k], W row-major [N x K]:
// lane n -> W row (n&15), K chunks at k0 + 8*(n>>4) and +16 more.
__device__ __forceinline__ v16bf frag_b(const bf16_t* base, int ldw, int lane) {
  const bf16_t* p = base + (size_t)(lane & 15) * ldw + ((lane >> 4) * 8);
  v8bf lo = *(const v8bf*)(p);
  v8bf hi = *(const v8bf*)(p + 16);
  return cat8(lo, hi);
}

__device__ __forceinline__ v8f wmma_bf16(v16bf a, v16bf b, v8f c) {
  return __builtin_amdgcn_wmma_f32_16x16x32_bf16(false, a, false, b,
                                                 (short)0, c, false, false);
}

__device__ __forceinline__ float sigmoidf_(float x) {
  return 1.0f / (1.0f + __expf(-x));
}

#if USE_TDM
// TDM: DMA a 2D tile (rows x cols bf16 elements, row stride in elements) from
// global to LDS at byte offset lds_off.  D# built per CDNA5 ISA:
//   group0: count=1, lds_addr, global_addr[56:0], type=2
//   group1: data_size=1 (2B), tensor_dim0/1, tile_dim0/1, tensor_dim0_stride
// Wave-level op (EXEC ignored); TENSORcnt tracks completion.
__device__ __forceinline__ void tdm_load_2d(const void* gptr, unsigned lds_off,
                                            unsigned rows, unsigned cols,
                                            unsigned row_stride) {
  unsigned long long ga = (unsigned long long)(size_t)gptr;
  u32x4 g0;
  g0[0] = 1u;                                         // count=1, user mode
  g0[1] = lds_off;                                    // lds_addr (bytes)
  g0[2] = (unsigned)(ga & 0xFFFFFFFFu);               // global_addr[31:0]
  g0[3] = (unsigned)((ga >> 32) & 0x01FFFFFFu) | (2u << 30);  // addr hi|type=2
  i32x8 g1;
  g1[0] = (int)(1u << 16);                            // data_size=1 (2 bytes)
  g1[1] = (int)((cols & 0xFFFFu) << 16);              // tensor_dim0[15:0]
  g1[2] = (int)(((cols >> 16) & 0xFFFFu) |            // tensor_dim0[31:16]
                ((rows & 0xFFFFu) << 16));            // tensor_dim1[15:0]
  g1[3] = (int)(((rows >> 16) & 0xFFFFu) |            // tensor_dim1[31:16]
                ((cols & 0xFFFFu) << 16));            // tile_dim0 = cols
  g1[4] = (int)(rows & 0xFFFFu);                      // tile_dim1 = rows
  g1[5] = (int)row_stride;                            // tensor_dim0_stride lo
  g1[6] = 0;
  g1[7] = 0;
  i32x4 z4 = {0, 0, 0, 0};
#if defined(__clang_major__) && __clang_major__ >= 23
  i32x8 z8 = {0, 0, 0, 0, 0, 0, 0, 0};
  __builtin_amdgcn_tensor_load_to_lds(g0, g1, z4, z4, z8, 0);
#else
  __builtin_amdgcn_tensor_load_to_lds(g0, g1, z4, z4, 0);
#endif
}
#endif

// ---------------------------------------------------------------- kernels ---

__global__ void k_embed_bf16(const int* __restrict__ seqs,
                             const float* __restrict__ embW,
                             bf16_t* __restrict__ emb) {
  int tok = seqs[blockIdx.x];
  const float* src = embW + (size_t)tok * EE;
  bf16_t* dst = emb + (size_t)blockIdx.x * EE;
  for (int i = threadIdx.x; i < EE; i += blockDim.x) dst[i] = f2bf(src[i]);
}

__global__ void k_f2bf(const float* __restrict__ src, bf16_t* __restrict__ dst,
                       int n) {
  for (int i = blockIdx.x * blockDim.x + threadIdx.x; i < n;
       i += gridDim.x * blockDim.x)
    dst[i] = f2bf(src[i]);
}

__global__ void k_init(float* hf32, bf16_t* hbf, unsigned int* cnt) {
  int i = blockIdx.x * blockDim.x + threadIdx.x;
  if (i < BB * HH2) { hf32[i] = 0.0f; hbf[i] = f2bf(0.0f); }
  if (i < 128) cnt[i] = 0u;
}

// xp[dir][BT,3H] = emb[BT,E] @ W_ih[dir]^T + b_ih[dir]
// grid (BT/128, 3H/64, 2), block 256 (8 waves); wave = 16 rows x 64 cols.
// W tile (64 rows x 512 K bf16 = 64KB) staged in LDS, shared by all 8 waves.
__global__ void __launch_bounds__(256)
k_gemm_xp(const bf16_t* __restrict__ emb,
          const bf16_t* __restrict__ W0, const bf16_t* __restrict__ W1,
          const float* __restrict__ b0,  const float* __restrict__ b1,
          float* __restrict__ xp0, float* __restrict__ xp1) {
  extern __shared__ char smem_raw[];
  bf16_t* sB = (bf16_t*)smem_raw;              // [64][EE] bf16 = 64KB

  const int lane = threadIdx.x & 31, wave = threadIdx.x >> 5;
  const int Mbase = blockIdx.x * 128 + wave * 16;
  const int Nbase = blockIdx.y * 64;
  const bf16_t* W  = blockIdx.z ? W1 : W0;
  const float* bia = blockIdx.z ? b1 : b0;
  float* xp        = blockIdx.z ? xp1 : xp0;

#if USE_TDM
  if (wave == 0) {
    tdm_load_2d(W + (size_t)Nbase * EE, (unsigned)(size_t)sB, 64, EE, EE);
    __builtin_amdgcn_s_wait_tensorcnt(0);
  }
#else
  for (int idx = threadIdx.x; idx < 64 * (EE / 8); idx += blockDim.x) {
    int rr = idx / (EE / 8), kk = (idx % (EE / 8)) * 8;
    *(v8bf*)(sB + (size_t)rr * EE + kk) =
        *(const v8bf*)(W + (size_t)(Nbase + rr) * EE + kk);
  }
#endif
  __syncthreads();

  v8f acc[4] = {v8f{}, v8f{}, v8f{}, v8f{}};
  for (int k0 = 0; k0 < EE; k0 += 32) {
    v16bf a = frag_a(emb + (size_t)Mbase * EE + k0, EE, lane);
#pragma unroll
    for (int c = 0; c < 4; ++c) {
      v16bf b = frag_b(sB + (size_t)(c * 16) * EE + k0, EE, lane);
      acc[c] = wmma_bf16(a, b, acc[c]);
    }
  }
  const int moff = Mbase + ((lane >> 4) * 8);
#pragma unroll
  for (int c = 0; c < 4; ++c) {
    int n = Nbase + c * 16 + (lane & 15);
    float bv = bia[n];
#pragma unroll
    for (int r = 0; r < 8; ++r)
      xp[(size_t)(moff + r) * HH3 + n] = acc[c][r] + bv;
  }
}

// Persistent recurrent scan.  grid 128 x 128 threads, dyn LDS 96KB.
// blockIdx: dir = bx>>6, hidden tile jt = bx&63.  wave w owns batch rows
// 16w..16w+15.  Per step: gh tiles (r,z,n) for (all B, 16 cols) via WMMA,
// gates, h update, then per-direction grid barrier.
__global__ void __launch_bounds__(128)
k_gru_scan(const float* __restrict__ xp0, const float* __restrict__ xp1,
           const bf16_t* __restrict__ Whh0, const bf16_t* __restrict__ Whh1,
           const float* __restrict__ bhh0, const float* __restrict__ bhh1,
           float* __restrict__ hf32, bf16_t* __restrict__ hbf,
           float* __restrict__ out, unsigned int* __restrict__ counters) {
  extern __shared__ char smem_raw[];
  bf16_t* sW = (bf16_t*)smem_raw;              // [3][16][HH] bf16 = 96KB

  const int lane = threadIdx.x & 31, wave = threadIdx.x >> 5;
  const int dir  = blockIdx.x >> 6;
  const int jt   = blockIdx.x & 63;
  const bf16_t* Whh = dir ? Whh1 : Whh0;
  const float*  xp  = dir ? xp1  : xp0;
  const float*  bhh = dir ? bhh1 : bhh0;
  unsigned int* cnt = counters + dir * 64;     // well-separated per direction

  // Stage this tile's W_hh slice into LDS: 3 gates x (16 rows x 1024 K).
#if USE_TDM
  if (wave == 0) {
#pragma unroll
    for (int g = 0; g < 3; ++g)
      tdm_load_2d(Whh + (size_t)(g * HH + jt * 16) * HH,
                  (unsigned)(size_t)sW + (unsigned)(g * 16 * HH * 2),
                  16, HH, HH);
    __builtin_amdgcn_s_wait_tensorcnt(0);
  }
#else
  const int nchunk = 3 * 16 * (HH / 8);
  for (int idx = threadIdx.x; idx < nchunk; idx += blockDim.x) {
    int g  = idx / (16 * (HH / 8));
    int rm = idx % (16 * (HH / 8));
    int rr = rm / (HH / 8);
    int kk = (rm % (HH / 8)) * 8;
    *(v8bf*)(sW + (size_t)(g * 16 + rr) * HH + kk) =
        *(const v8bf*)(Whh + (size_t)(g * HH + jt * 16 + rr) * HH + kk);
  }
#endif
  __syncthreads();

  const int b0 = wave * 16;
  const int j  = jt * 16 + (lane & 15);
  const float br_ = bhh[j], bz_ = bhh[HH + j], bn_ = bhh[2 * HH + j];
  const size_t hcol = (size_t)dir * HH + j;

  for (int step = 0; step < TT; ++step) {
    const int t = dir ? (TT - 1 - step) : step;

    v8f ar = v8f{}, az = v8f{}, an = v8f{};
    const bf16_t* hrow = hbf + (size_t)b0 * HH2 + (size_t)dir * HH;
    for (int k0 = 0; k0 < HH; k0 += 32) {
      v16bf a  = frag_a(hrow + k0, HH2, lane);
      v16bf br = frag_b(sW + (size_t)(0 * 16) * HH + k0, HH, lane);
      v16bf bz = frag_b(sW + (size_t)(1 * 16) * HH + k0, HH, lane);
      v16bf bn = frag_b(sW + (size_t)(2 * 16) * HH + k0, HH, lane);
      ar = wmma_bf16(a, br, ar);
      az = wmma_bf16(a, bz, az);
      an = wmma_bf16(a, bn, an);
    }

#pragma unroll
    for (int r = 0; r < 8; ++r) {
      int b = b0 + r + ((lane >> 4) * 8);
      size_t xb = ((size_t)b * TT + t) * HH3;
      float xr = xp[xb + j], xz = xp[xb + HH + j], xn = xp[xb + 2 * HH + j];
      float hp = hf32[(size_t)b * HH2 + hcol];
      float rg = sigmoidf_(xr + ar[r] + br_);
      float zg = sigmoidf_(xz + az[r] + bz_);
      float ng = tanhf(xn + rg * (an[r] + bn_));
      float hn = (1.0f - zg) * ng + zg * hp;
      hf32[(size_t)b * HH2 + hcol] = hn;
      hbf [(size_t)b * HH2 + hcol] = f2bf(hn);
      out[((size_t)b * TT + t) * HH2 + hcol] = hn;
    }

    // grid barrier over this direction's 64 workgroups
    __threadfence();
    __syncthreads();
    if (threadIdx.x == 0) {
      __hip_atomic_fetch_add(cnt, 1u, __ATOMIC_ACQ_REL,
                             __HIP_MEMORY_SCOPE_AGENT);
      unsigned target = (unsigned)(step + 1) * 64u;
      while (__hip_atomic_load(cnt, __ATOMIC_ACQUIRE,
                               __HIP_MEMORY_SCOPE_AGENT) < target)
        __builtin_amdgcn_s_sleep(2);
    }
    __syncthreads();
    __threadfence();
  }
}

// hiddens = tanh(hcat[B,2H] @ fc_W^T + fc_b).  grid 16 x 128 (4 waves=M tiles)
__global__ void __launch_bounds__(128)
k_fc(const bf16_t* __restrict__ hbf, const bf16_t* __restrict__ fcW,
     const float* __restrict__ fcb, float* __restrict__ outh) {
  const int lane = threadIdx.x & 31, wave = threadIdx.x >> 5;
  const int Nbase = blockIdx.x * 64;
  const int Mbase = wave * 16;
  v8f acc[4] = {v8f{}, v8f{}, v8f{}, v8f{}};
  for (int k0 = 0; k0 < HH2; k0 += 32) {
    v16bf a = frag_a(hbf + (size_t)Mbase * HH2 + k0, HH2, lane);
#pragma unroll
    for (int c = 0; c < 4; ++c) {
      v16bf b = frag_b(fcW + (size_t)(Nbase + c * 16) * HH2 + k0, HH2, lane);
      acc[c] = wmma_bf16(a, b, acc[c]);
    }
  }
#pragma unroll
  for (int c = 0; c < 4; ++c) {
    int n = Nbase + c * 16 + (lane & 15);
    float bv = fcb[n];
#pragma unroll
    for (int r = 0; r < 8; ++r) {
      int b = Mbase + r + ((lane >> 4) * 8);
      outh[(size_t)b * HH + n] = tanhf(acc[c][r] + bv);
    }
  }
}

// ------------------------------------------------------------------ host ---

extern "C" void kernel_launch(void* const* d_in, const int* in_sizes, int n_in,
                              void* d_out, int out_size, void* d_ws,
                              size_t ws_size, hipStream_t stream) {
  const int*   seqs  = (const int*)  d_in[0];
  const float* embW  = (const float*)d_in[1];
  const float* Wih_f = (const float*)d_in[2];
  const float* Whh_f = (const float*)d_in[3];
  const float* bih_f = (const float*)d_in[4];
  const float* bhh_f = (const float*)d_in[5];
  const float* Wih_b = (const float*)d_in[6];
  const float* Whh_b = (const float*)d_in[7];
  const float* bih_b = (const float*)d_in[8];
  const float* bhh_b = (const float*)d_in[9];
  const float* fcW   = (const float*)d_in[10];
  const float* fcb   = (const float*)d_in[11];
  float* out = (float*)d_out;

  char* ws = (char*)d_ws;
  size_t off = 0;
  bf16_t* emb_bf  = (bf16_t*)(ws + off); off += (size_t)BT * EE * 2;
  bf16_t* WihF_bf = (bf16_t*)(ws + off); off += (size_t)HH3 * EE * 2;
  bf16_t* WihB_bf = (bf16_t*)(ws + off); off += (size_t)HH3 * EE * 2;
  bf16_t* WhhF_bf = (bf16_t*)(ws + off); off += (size_t)HH3 * HH * 2;
  bf16_t* WhhB_bf = (bf16_t*)(ws + off); off += (size_t)HH3 * HH * 2;
  bf16_t* fcW_bf  = (bf16_t*)(ws + off); off += (size_t)HH * HH2 * 2;
  float*  xp0     = (float*) (ws + off); off += (size_t)BT * HH3 * 4;
  float*  xp1     = (float*) (ws + off); off += (size_t)BT * HH3 * 4;
  float*  hf32    = (float*) (ws + off); off += (size_t)BB * HH2 * 4;
  bf16_t* hbf     = (bf16_t*)(ws + off); off += (size_t)BB * HH2 * 2;
  unsigned int* counters = (unsigned int*)(ws + off); off += 512;

  // 1) embedding gather -> bf16
  k_embed_bf16<<<BT, 128, 0, stream>>>(seqs, embW, emb_bf);

  // 2) weight conversions -> bf16
  k_f2bf<<<1536, 256, 0, stream>>>(Wih_f, WihF_bf, HH3 * EE);
  k_f2bf<<<1536, 256, 0, stream>>>(Wih_b, WihB_bf, HH3 * EE);
  k_f2bf<<<3072, 256, 0, stream>>>(Whh_f, WhhF_bf, HH3 * HH);
  k_f2bf<<<3072, 256, 0, stream>>>(Whh_b, WhhB_bf, HH3 * HH);
  k_f2bf<<<2048, 256, 0, stream>>>(fcW,   fcW_bf,  HH * HH2);

  // 3) zero h state + barrier counters (every launch: determinism)
  k_init<<<(BB * HH2 + 255) / 256, 256, 0, stream>>>(hf32, hbf, counters);

  // 4) input projections (both directions), 64KB dynamic LDS
  (void)hipFuncSetAttribute((const void*)k_gemm_xp,
                            hipFuncAttributeMaxDynamicSharedMemorySize, 65536);
  k_gemm_xp<<<dim3(BT / 128, HH3 / 64, 2), 256, 65536, stream>>>(
      emb_bf, WihF_bf, WihB_bf, bih_f, bih_b, xp0, xp1);

  // 5) persistent bidirectional scan (96KB dynamic LDS per WG)
  (void)hipFuncSetAttribute((const void*)k_gru_scan,
                            hipFuncAttributeMaxDynamicSharedMemorySize, 98304);
  k_gru_scan<<<128, 128, 98304, stream>>>(xp0, xp1, WhhF_bf, WhhB_bf,
                                          bhh_f, bhh_b, hf32, hbf, out,
                                          counters);

  // 6) final FC on concatenated last hidden states
  k_fc<<<HH / 64, 128, 0, stream>>>(hbf, fcW_bf, fcb,
                                    out + (size_t)BT * HH2);
}
